// GatedMultimodalLayerGNN_43739946943048
// MI455X (gfx1250) — matrix operand order repaired and checked by reading, not compiled, not run on hardware
//
#include <hip/hip_runtime.h>

// CDNA5 / gfx1250 implementation of the 4-node-GCN over 1M rows.
//
// Math: rows >=4 are a pure per-row MLP: out = relu(x W1^T + b1) W2^T + b2.
// Rows 0..3 all equal relu(0.25*sum_{j<4} x_j W1^T + b1) W2^T + b2 (fixup kernel).
//
// Precision: bf16x3 split (Ahi*Bhi + Ahi*Blo + Alo*Bhi) with f32 WMMA
// accumulators => ~fp32 accuracy at bf16 matrix-core rates.

typedef __attribute__((ext_vector_type(16))) __bf16 v16bf;
typedef __attribute__((ext_vector_type(8)))  float  v8f;
typedef __attribute__((ext_vector_type(4)))  float  v4f;

#define WAVES_PER_BLOCK 4
#define TILES_PER_WAVE  8
#define HPITCH          68   // 64 + 4 pad (keeps 16B alignment, breaks bank conflicts)

static __device__ __forceinline__ v8f wmma_bf16(v16bf a, v16bf b, v8f c) {
  // (neg_a, A, neg_b, B, c_mod, C, reuse_a, reuse_b)
  return __builtin_amdgcn_wmma_f32_16x16x32_bf16(false, a, false, b, (short)0, c,
                                                 false, false);
}

// Split 16 f32 values into bf16 hi + bf16 residual lo (A-fragment element order).
static __device__ __forceinline__ void split_bf16x2(v4f a0, v4f a1, v4f a2, v4f a3,
                                                    v16bf& hi, v16bf& lo) {
  float xs[16];
#pragma unroll
  for (int i = 0; i < 4; ++i) {
    xs[i] = a0[i]; xs[4 + i] = a1[i]; xs[8 + i] = a2[i]; xs[12 + i] = a3[i];
  }
#pragma unroll
  for (int e = 0; e < 16; ++e) {
    __bf16 h = (__bf16)xs[e];
    hi[e] = h;
    lo[e] = (__bf16)(xs[e] - (float)h);
  }
}

__global__ void __launch_bounds__(WAVES_PER_BLOCK * 32)
gcn_mlp_main(const float* __restrict__ audio, const float* __restrict__ video,
             const float* __restrict__ x1in,  const float* __restrict__ x2in,
             const float* __restrict__ W1, const float* __restrict__ b1,
             const float* __restrict__ W2, const float* __restrict__ b2,
             float* __restrict__ out, int n)
{
  // Weight fragments pre-split to bf16 hi/lo, stored in WMMA B-fragment order:
  //   frag index [hi/lo][ntile][ktile][lane][elem], elem e -> K = (e/8)*16+(lane/16)*8+e%8,
  //   column N = ntile*16 + lane%16.  Each lane's 16 elems are 32B contiguous.
  __shared__ __attribute__((aligned(32))) __bf16 w1f[2][4][4][32][16]; // 32 KB
  __shared__ __attribute__((aligned(32))) __bf16 w2f[2][4][2][32][16]; // 16 KB
  __shared__ __attribute__((aligned(16))) float  hst[WAVES_PER_BLOCK][16][HPITCH];

  const int tid = threadIdx.x;

  // ---- stage + split weights into LDS (B-fragment layout) ----
  for (int f = tid; f < 4 * 4 * 32 * 16; f += blockDim.x) {
    int e    = f & 15;
    int lane = (f >> 4) & 31;
    int kt   = (f >> 9) & 3;
    int nt   = (f >> 11) & 3;
    int o = nt * 16 + (lane & 15);
    int k = kt * 32 + (e >> 3) * 16 + (lane >> 4) * 8 + (e & 7);
    float v = W1[o * 128 + k];
    __bf16 h = (__bf16)v;
    w1f[0][nt][kt][lane][e] = h;
    w1f[1][nt][kt][lane][e] = (__bf16)(v - (float)h);
  }
  for (int f = tid; f < 4 * 2 * 32 * 16; f += blockDim.x) {
    int e    = f & 15;
    int lane = (f >> 4) & 31;
    int kt   = (f >> 9) & 1;
    int nt   = (f >> 10) & 3;
    int o = nt * 16 + (lane & 15);
    int k = kt * 32 + (e >> 3) * 16 + (lane >> 4) * 8 + (e & 7);
    float v = W2[o * 64 + k];
    __bf16 h = (__bf16)v;
    w2f[0][nt][kt][lane][e] = h;
    w2f[1][nt][kt][lane][e] = (__bf16)(v - (float)h);
  }
  __syncthreads();

  const int lane  = tid & 31;
  const int wave  = tid >> 5;
  const int lrow  = lane & 15;  // A-frag row / C-frag column (within tile)
  const int lhalf = lane >> 4;  // half-wave group

  float b1v[4], b2v[4];
#pragma unroll
  for (int nt = 0; nt < 4; ++nt) {
    b1v[nt] = b1[nt * 16 + lrow];
    b2v[nt] = b2[nt * 16 + lrow];
  }

  const float* __restrict__ ins[4] = {audio, video, x1in, x2in};
  const int ntiles  = n >> 4;
  const int tstride = gridDim.x * WAVES_PER_BLOCK;

  for (int t = blockIdx.x * WAVES_PER_BLOCK + wave; t < ntiles; t += tstride) {
    const int rowbase = t << 4;
    const int arow    = rowbase + lrow;

    // -------- layer 1: [16x128] x [128x64] --------
    v8f zero = {};
    v8f acc[4] = {zero, zero, zero, zero};
#pragma unroll
    for (int kt = 0; kt < 4; ++kt) {
      // A-fragment: lane reads K(e) = kt*32 + (e/8)*16 + lhalf*8 + e%8;
      // two 32B runs; half-wave pair consumes each 128B input row fully.
      const v4f* p = (const v4f*)(ins[kt] + (size_t)arow * 32);
      const int q = lhalf * 2;
      v4f a0 = __builtin_nontemporal_load(p + q);
      v4f a1 = __builtin_nontemporal_load(p + q + 1);
      v4f a2 = __builtin_nontemporal_load(p + q + 4);
      v4f a3 = __builtin_nontemporal_load(p + q + 5);
      v16bf ahi, alo;
      split_bf16x2(a0, a1, a2, a3, ahi, alo);
#pragma unroll
      for (int nt = 0; nt < 4; ++nt) {
        v16bf bhi = *(const v16bf*)&w1f[0][nt][kt][lane][0];
        v16bf blo = *(const v16bf*)&w1f[1][nt][kt][lane][0];
        acc[nt] = wmma_bf16(ahi, bhi, acc[nt]);
        acc[nt] = wmma_bf16(ahi, blo, acc[nt]);
        acc[nt] = wmma_bf16(alo, bhi, acc[nt]);
      }
    }

    // bias + relu, stage C-fragment layout -> LDS tile [row][col]
#pragma unroll
    for (int nt = 0; nt < 4; ++nt) {
#pragma unroll
      for (int v = 0; v < 8; ++v) {
        float g = acc[nt][v] + b1v[nt];
        hst[wave][lhalf * 8 + v][nt * 16 + lrow] = g > 0.f ? g : 0.f;
      }
    }
    // per-wave tile: same-wave DS ordering, just wait for the stores to land
    asm volatile("s_wait_dscnt 0" ::: "memory");

    // -------- layer 2: [16x64] x [64x64] --------
    v8f acc2[4] = {zero, zero, zero, zero};
#pragma unroll
    for (int kt = 0; kt < 2; ++kt) {
      const float* hb = &hst[wave][lrow][kt * 32 + lhalf * 8];
      v4f a0 = *(const v4f*)(hb);
      v4f a1 = *(const v4f*)(hb + 4);
      v4f a2 = *(const v4f*)(hb + 16);
      v4f a3 = *(const v4f*)(hb + 20);
      v16bf ahi, alo;
      split_bf16x2(a0, a1, a2, a3, ahi, alo);
#pragma unroll
      for (int nt = 0; nt < 4; ++nt) {
        v16bf bhi = *(const v16bf*)&w2f[0][nt][kt][lane][0];
        v16bf blo = *(const v16bf*)&w2f[1][nt][kt][lane][0];
        acc2[nt] = wmma_bf16(ahi, bhi, acc2[nt]);
        acc2[nt] = wmma_bf16(ahi, blo, acc2[nt]);
        acc2[nt] = wmma_bf16(alo, bhi, acc2[nt]);
      }
    }

    // bias + streaming (non-temporal) store; per instruction each half-wave
    // writes a contiguous 64B row segment.
#pragma unroll
    for (int nt = 0; nt < 4; ++nt) {
#pragma unroll
      for (int v = 0; v < 8; ++v) {
        float r = acc2[nt][v] + b2v[nt];
        __builtin_nontemporal_store(
            r, &out[(size_t)(rowbase + lhalf * 8 + v) * 64 + nt * 16 + lrow]);
      }
    }
  }
}

// Exact fp32 fixup for rows 0..3 (the only rows touched by the 4-node graph):
// all four equal relu(0.25*sum_{j<4} x_j W1^T + b1) W2^T + b2.
__global__ void __launch_bounds__(64)
gcn_fixup(const float* __restrict__ audio, const float* __restrict__ video,
          const float* __restrict__ x1in,  const float* __restrict__ x2in,
          const float* __restrict__ W1, const float* __restrict__ b1,
          const float* __restrict__ W2, const float* __restrict__ b2,
          float* __restrict__ out)
{
  __shared__ float hh[64];
  const int o = threadIdx.x;  // 64 threads, one per output feature
  const float* ins[4] = {audio, video, x1in, x2in};
  float s = 0.f;
  for (int j = 0; j < 4; ++j)
    for (int c = 0; c < 4; ++c)
      for (int k = 0; k < 32; ++k)
        s += ins[c][j * 32 + k] * W1[o * 128 + c * 32 + k];
  float h = 0.25f * s + b1[o];
  hh[o] = h > 0.f ? h : 0.f;
  __syncthreads();
  float s2 = 0.f;
  for (int k = 0; k < 64; ++k) s2 += hh[k] * W2[o * 64 + k];
  float r = s2 + b2[o];
  for (int j = 0; j < 4; ++j) out[j * 64 + o] = r;
}

extern "C" void kernel_launch(void* const* d_in, const int* in_sizes, int n_in,
                              void* d_out, int out_size, void* d_ws, size_t ws_size,
                              hipStream_t stream) {
  (void)n_in; (void)out_size; (void)d_ws; (void)ws_size;
  const float* audio = (const float*)d_in[0];
  const float* video = (const float*)d_in[1];
  const float* x1in  = (const float*)d_in[2];
  const float* x2in  = (const float*)d_in[3];
  const float* W1    = (const float*)d_in[4];
  const float* b1    = (const float*)d_in[5];
  const float* W2    = (const float*)d_in[6];
  const float* b2    = (const float*)d_in[7];
  float* out = (float*)d_out;

  const int n     = in_sizes[0] / 32;  // N rows
  const int tiles = n >> 4;
  int blocks = (tiles + WAVES_PER_BLOCK * TILES_PER_WAVE - 1) /
               (WAVES_PER_BLOCK * TILES_PER_WAVE);
  if (blocks < 1) blocks = 1;

  gcn_mlp_main<<<blocks, WAVES_PER_BLOCK * 32, 0, stream>>>(
      audio, video, x1in, x2in, W1, b1, W2, b2, out, n);
  if (n >= 4)
    gcn_fixup<<<1, 64, 0, stream>>>(audio, video, x1in, x2in, W1, b1, W2, b2, out);
}
